// SimpleEmbeddingNetwork_20220706029768
// MI455X (gfx1250) — compile-verified, matrix operand order checked by reading, not run on hardware
//
#include <hip/hip_runtime.h>
#include <math.h>

#define N_NODES 8192
#define KNN 16
#define IDIM 5
#define HDIM 64
#define MDIM 96
#define ODIM 8
#define NEDGE (N_NODES * KNN)
#define EPSBN 1e-5f

typedef __attribute__((ext_vector_type(16))) _Float16 v16h;
typedef __attribute__((ext_vector_type(8)))  _Float16 v8h;
typedef __attribute__((ext_vector_type(8)))  float    v8f;

#define ACT_NONE 0
#define ACT_ELU  1
#define ACT_TANH 2

__device__ __forceinline__ float act_apply(float v, int act) {
  if (act == ACT_ELU)  return v > 0.f ? v : (expf(v) - 1.f);
  if (act == ACT_TANH) return tanhf(v);
  return v;
}

__device__ __forceinline__ v16h cat16(v8h lo, v8h hi) {
  return __builtin_shufflevector(lo, hi, 0, 1, 2, 3, 4, 5, 6, 7,
                                 8, 9, 10, 11, 12, 13, 14, 15);
}

// ---------------------------------------------------------------------------
// LDS-free GEMM: C[M x Nc] = act(A[M x K] @ W + bias)
// A: f16, K-major, ldk = Kpad (K % 32 == 0, zero padded).
// Wt: f16, TRANSPOSED [Nc x Kpad] (prep_wt_kernel), L2/WGP$-resident.
// Lane fragment = two contiguous 16B runs at k offsets kb and 16+kb
// (CDNA5 ISA 7.12.2) -> global_load_b128 pairs, no staging, no LDS.
// NT = #16-col tiles is a template parameter: branch-free inner loop, all
// B loads issued in one clause, then NT back-to-back WMMAs.
// M % 64 == 0.
// ---------------------------------------------------------------------------
#define GW 4
template <int NT>
__global__ void gemm_h_kernel(const _Float16* __restrict__ A, int ldk,
                              const _Float16* __restrict__ Wt,
                              const float* __restrict__ bias,
                              float* __restrict__ C32,
                              _Float16* __restrict__ C16, int ldc,
                              int Kdim, int Ncols, int act)
{
  const int wave = threadIdx.x >> 5;
  const int lane = threadIdx.x & 31;
  const int row0 = (blockIdx.x * GW + wave) * 16;
  const int rA = lane & 15;
  const int kb = (lane >= 16) ? 8 : 0;

  v8f acc[NT];
#pragma unroll
  for (int c = 0; c < NT; ++c) acc[c] = (v8f){0.f,0.f,0.f,0.f,0.f,0.f,0.f,0.f};

  const _Float16* ap  = &A[(size_t)(row0 + rA) * ldk + kb];
  const _Float16* wp0 = &Wt[(size_t)rA * ldk + kb];

  for (int k0 = 0; k0 < Kdim; k0 += 32) {
    v16h af = cat16(*(const v8h*)(ap + k0), *(const v8h*)(ap + k0 + 16));
    __builtin_prefetch(ap + k0 + 32, 0, 1);    // speculative, safe past end
    v16h bf[NT];
#pragma unroll
    for (int c = 0; c < NT; ++c) {
      const _Float16* wp = wp0 + (size_t)c * 16 * ldk + k0;
      bf[c] = cat16(*(const v8h*)wp, *(const v8h*)(wp + 16));
    }
#pragma unroll
    for (int c = 0; c < NT; ++c)
      acc[c] = __builtin_amdgcn_wmma_f32_16x16x32_f16(false, af, false, bf[c],
                                                      (short)0, acc[c],
                                                      false, false);
  }

  const int rb0 = row0 + ((lane >= 16) ? 8 : 0);
#pragma unroll
  for (int c = 0; c < NT; ++c) {
    int cn = c * 16 + (lane & 15);
    if (cn < Ncols) {
      float bb = bias[cn];
#pragma unroll
      for (int j = 0; j < 8; ++j) {
        float v = act_apply(acc[c][j] + bb, act);
        if (C32) C32[(size_t)(rb0 + j) * ldc + cn] = v;
        if (C16) C16[(size_t)(rb0 + j) * ldc + cn] = (_Float16)v;
      }
    }
  }
}

// Wt[c*Kpad + k] = (k < K) ? (f16)W[k*Nc + c] : 0
__global__ void prep_wt_kernel(const float* __restrict__ W, int K, int Nc,
                               int Kpad, _Float16* __restrict__ Wt) {
  int i = blockIdx.x * blockDim.x + threadIdx.x;
  if (i >= Nc * Kpad) return;
  int c = i / Kpad, k = i % Kpad;
  Wt[i] = (k < K) ? (_Float16)W[(size_t)k * Nc + c] : (_Float16)0.f;
}

// ---------------------------------------------------------------------------
// kNN on an f16 [Nn x 64] feature image (zero-padded K). Fragments loaded
// directly from global (L2-resident); only the 16x16 distance tile uses LDS.
// ---------------------------------------------------------------------------
__global__ void knn_kernel(const _Float16* __restrict__ H16,
                           const float* __restrict__ sq,
                           int* __restrict__ knn, int Nn)
{
  __shared__ float Dt[16 * 16];
  const int lane = threadIdx.x;
  const int row0 = blockIdx.x * 16;
  const int rA = lane & 15;
  const int kb = (lane >= 16) ? 8 : 0;

  const _Float16* apr = &H16[(size_t)(row0 + rA) * 64 + kb];
  v16h af0 = cat16(*(const v8h*)(apr +  0), *(const v8h*)(apr + 16));
  v16h af1 = cat16(*(const v8h*)(apr + 32), *(const v8h*)(apr + 48));

  const int rb = (lane >= 16) ? 8 : 0;
  float sqa[8];
#pragma unroll
  for (int j = 0; j < 8; ++j) sqa[j] = sq[row0 + rb + j];

  float bestd[KNN]; int besti[KNN];
#pragma unroll
  for (int t = 0; t < KNN; ++t) { bestd[t] = __builtin_inff(); besti[t] = 0; }

  for (int c0 = 0; c0 < Nn; c0 += 16) {
    const _Float16* bpr = &H16[(size_t)(c0 + rA) * 64 + kb];
    v16h bf0 = cat16(*(const v8h*)(bpr +  0), *(const v8h*)(bpr + 16));
    v16h bf1 = cat16(*(const v8h*)(bpr + 32), *(const v8h*)(bpr + 48));
    v8f acc = {0.f,0.f,0.f,0.f,0.f,0.f,0.f,0.f};
    acc = __builtin_amdgcn_wmma_f32_16x16x32_f16(false, af0, false, bf0,
                                                 (short)0, acc, false, false);
    acc = __builtin_amdgcn_wmma_f32_16x16x32_f16(false, af1, false, bf1,
                                                 (short)0, acc, false, false);
    float sqb = sq[c0 + rA];
#pragma unroll
    for (int j = 0; j < 8; ++j) {
      float d = sqa[j] - 2.f * acc[j] + sqb;
      if (row0 + rb + j == c0 + rA) d = __builtin_inff();
      Dt[(rb + j) * 16 + rA] = d;
    }
    if (lane < 16) {
      for (int n = 0; n < 16; ++n) {
        float nd = Dt[lane * 16 + n];
        int   ni = c0 + n;
        if (nd < bestd[KNN - 1]) {
#pragma unroll
          for (int t = 0; t < KNN; ++t) {
            if (nd < bestd[t]) {
              float od = bestd[t]; int oi = besti[t];
              bestd[t] = nd; besti[t] = ni;
              nd = od; ni = oi;
            }
          }
        }
      }
    }
  }
  if (lane < 16) {
#pragma unroll
    for (int t = 0; t < KNN; ++t)
      knn[(size_t)(row0 + lane) * KNN + t] = besti[t];
  }
}

// ---------------------------------------------------------------------------
// f32 -> f16 converters
// ---------------------------------------------------------------------------
__global__ void tof16_kernel(const float* __restrict__ src, int rows, int F,
                             int ldd, _Float16* __restrict__ dst) {
  int i = blockIdx.x * blockDim.x + threadIdx.x;
  if (i >= rows * ldd) return;
  int c = i % ldd, r = i / ldd;
  dst[i] = (c < F) ? (_Float16)src[(size_t)r * F + c] : (_Float16)0.f;
}
__global__ void pad_x16_kernel(const float* __restrict__ x,
                               _Float16* __restrict__ xp) {
  int i = blockIdx.x * blockDim.x + threadIdx.x;
  if (i >= N_NODES * 32) return;
  int c = i & 31, n = i >> 5;
  xp[i] = (c < IDIM) ? (_Float16)x[n * IDIM + c] : (_Float16)0.f;
}

// ---------------------------------------------------------------------------
// Elementwise / reduction helpers
// ---------------------------------------------------------------------------
__global__ void rownorm_kernel(const float* __restrict__ Hf, int F,
                               float* __restrict__ sq, int Nn) {
  int i = blockIdx.x * blockDim.x + threadIdx.x;
  if (i >= Nn) return;
  float s = 0.f;
  for (int k = 0; k < F; ++k) { float v = Hf[(size_t)i * F + k]; s += v * v; }
  sq[i] = s;
}

__global__ void zero_f32_kernel(float* p, int n) {
  int i = blockIdx.x * blockDim.x + threadIdx.x;
  if (i < n) p[i] = 0.f;
}

__global__ void bn_stats_kernel(const float* __restrict__ Hm, int ld, int rows,
                                int cols, const float* __restrict__ wgt,
                                float* __restrict__ stats, int rpb) {
  int c = threadIdx.x;
  int r0 = blockIdx.x * rpb;
  int r1 = min(r0 + rpb, rows);
  float s = 0.f, s2 = 0.f;
  for (int r = r0; r < r1; ++r) {
    float v = Hm[(size_t)r * ld + c];
    float w = wgt ? wgt[r] : 1.f;
    s += w * v; s2 += w * v * v;
  }
  atomicAdd(&stats[c], s);
  atomicAdd(&stats[cols + c], s2);
  if (wgt && c == 0) {
    float ws = 0.f;
    for (int r = r0; r < r1; ++r) ws += wgt[r];
    atomicAdd(&stats[2 * cols], ws);
  }
}

// In-place BN (+activation); optional f16 mirror for feeding the next GEMM.
__global__ void bn_apply_kernel(float* __restrict__ Hm, int ld, int rows,
                                int cols, const float* __restrict__ g,
                                const float* __restrict__ b,
                                const float* __restrict__ stats,
                                float cntIn, int act,
                                _Float16* __restrict__ out16) {
  int idx = blockIdx.x * blockDim.x + threadIdx.x;
  if (idx >= rows * cols) return;
  int r = idx / cols, c = idx % cols;
  float cnt = cntIn > 0.f ? cntIn : stats[2 * cols];
  float mu  = stats[c] / cnt;
  float var = stats[cols + c] / cnt - mu * mu;
  float v = act_apply(
      g[c] * (Hm[(size_t)r * ld + c] - mu) * rsqrtf(var + EPSBN) + b[c], act);
  Hm[(size_t)r * ld + c] = v;
  if (out16) out16[(size_t)r * cols + c] = (_Float16)v;
}

// mode 0: [f_tgt, f_src - f_tgt]; mode 1: [f_src, f_tgt]; f16 out, zero pad.
__global__ void edge_gather_kernel(const float* __restrict__ f, int F,
                                   const int* __restrict__ srcIdx, int mode,
                                   _Float16* __restrict__ Aed, int ldo) {
  int e = blockIdx.x * blockDim.x + threadIdx.x;
  if (e >= NEDGE) return;
  int t = e / KNN;
  int s = srcIdx[e];
  _Float16* o = Aed + (size_t)e * ldo;
  if (mode == 0) {
    for (int c = 0; c < F; ++c) {
      float xi = f[(size_t)t * F + c], xj = f[(size_t)s * F + c];
      o[c] = (_Float16)xi; o[F + c] = (_Float16)(xj - xi);
    }
  } else {
    for (int c = 0; c < F; ++c) {
      o[c] = (_Float16)f[(size_t)s * F + c];
      o[F + c] = (_Float16)f[(size_t)t * F + c];
    }
  }
  for (int c = 2 * F; c < ldo; ++c) o[c] = (_Float16)0.f;
}

__global__ void concat3_kernel(const float* a, const float* b2, const float* c,
                               float* out) {  // [H | H | I]
  int i = blockIdx.x * blockDim.x + threadIdx.x;
  if (i >= N_NODES) return;
  float* o = out + (size_t)i * (2 * HDIM + IDIM);
  for (int k = 0; k < HDIM; ++k) o[k] = a[(size_t)i * HDIM + k];
  for (int k = 0; k < HDIM; ++k) o[HDIM + k] = b2[(size_t)i * HDIM + k];
  for (int k = 0; k < IDIM; ++k) o[2 * HDIM + k] = c[(size_t)i * IDIM + k];
}

__global__ void scatter_sum_res_kernel(float* __restrict__ node,
                                       const float* __restrict__ m, int ldm) {
  int idx = blockIdx.x * blockDim.x + threadIdx.x;
  if (idx >= N_NODES * HDIM) return;
  int n = idx / HDIM, c = idx % HDIM;
  float s = 0.f;
  for (int k = 0; k < KNN; ++k) s += m[(size_t)(n * KNN + k) * ldm + c];
  node[idx] += s;
}

__global__ void prop_max_res_kernel(float* __restrict__ node,
                                    const float* __restrict__ m, int ldm,
                                    const float* __restrict__ w) {
  int idx = blockIdx.x * blockDim.x + threadIdx.x;
  if (idx >= N_NODES * HDIM) return;
  int n = idx / HDIM, c = idx % HDIM;
  float mx = -__builtin_inff();
  for (int k = 0; k < KNN; ++k) {
    int e = n * KNN + k;
    if (w[e] > 0.f) mx = fmaxf(mx, m[(size_t)e * ldm + c]);
  }
  node[idx] += (mx > -3.0e38f) ? mx : 0.f;
}

__global__ void add2_kernel(float* dst, const float* a, const float* b, int n) {
  int i = blockIdx.x * blockDim.x + threadIdx.x;
  if (i < n) dst[i] = a[i] + b[i];
}

__global__ void good_kernel(const float* __restrict__ scores,
                            const int* __restrict__ col,
                            float* __restrict__ wv, int* __restrict__ srcU,
                            float* __restrict__ eix) {
  int e = blockIdx.x * blockDim.x + threadIdx.x;
  if (e >= NEDGE) return;
  float s0 = scores[2 * e], s1 = scores[2 * e + 1];
  int g = (s1 > s0) ? 1 : 0;
  int t = e / KNN;
  wv[e] = g ? 1.f : 0.f;
  srcU[e] = g ? col[e] : t;
  eix[e] = (float)col[e];
  eix[NEDGE + e] = (float)t;
}

// ---- connected components: min-label propagation + pointer jumping ----
__global__ void init_labels_kernel(int* l, int n) {
  int i = blockIdx.x * blockDim.x + threadIdx.x;
  if (i < n) l[i] = i;
}
__global__ void relax_kernel(int* __restrict__ l, const int* __restrict__ srcU) {
  int e = blockIdx.x * blockDim.x + threadIdx.x;
  if (e >= NEDGE) return;
  int t = e / KNN, s = srcU[e];
  int ls = l[s], lt = l[t];
  int m = ls < lt ? ls : lt;
  atomicMin(&l[s], m);
  atomicMin(&l[t], m);
}
__global__ void jump_kernel(const int* __restrict__ l, int* __restrict__ lo, int n) {
  int i = blockIdx.x * blockDim.x + threadIdx.x;
  if (i < n) lo[i] = l[l[i]];
}
__global__ void mark_kernel(const int* l, int* present, int n) {
  int i = blockIdx.x * blockDim.x + threadIdx.x;
  if (i < n) present[l[i]] = 1;
}
__global__ void scan_kernel(const int* present, int* cum, int n) {
  if (blockIdx.x == 0 && threadIdx.x == 0) {
    int c = 0;
    for (int i = 0; i < n; ++i) { c += present[i]; cum[i] = c; }
  }
}
__global__ void cmap_kernel(const int* l, const int* cum, int* cmap,
                            float* outf, int n) {
  int i = blockIdx.x * blockDim.x + threadIdx.x;
  if (i >= n) return;
  int cm = cum[l[i]] - 1;
  cmap[i] = cm;
  outf[i] = (float)cm;
}

// ---- cluster max-pool via order-preserving uint encoding ----
__device__ __forceinline__ unsigned fenc(float f) {
  unsigned u = __float_as_uint(f);
  return (u & 0x80000000u) ? ~u : (u | 0x80000000u);
}
__device__ __forceinline__ float fdec(unsigned e) {
  unsigned u = (e & 0x80000000u) ? (e & 0x7FFFFFFFu) : ~e;
  return __uint_as_float(u);
}
__global__ void encinit_kernel(unsigned* enc, int n) {
  int i = blockIdx.x * blockDim.x + threadIdx.x;
  if (i < n) enc[i] = 0x007FFFFFu;  // fenc(-inf)
}
__global__ void pool_scatter_kernel(const float* __restrict__ xp,
                                    const int* __restrict__ cmap,
                                    unsigned* __restrict__ enc, int n) {
  int idx = blockIdx.x * blockDim.x + threadIdx.x;
  if (idx >= n) return;
  int nd = idx / HDIM;
  int c = idx % HDIM;
  atomicMax(&enc[(size_t)cmap[nd] * HDIM + c], fenc(xp[idx]));
}
__global__ void pool_decode_kernel(const unsigned* enc, float* pooled, int n) {
  int i = blockIdx.x * blockDim.x + threadIdx.x;
  if (i >= n) return;
  float v = fdec(enc[i]);
  pooled[i] = (v > -3.0e38f) ? v : 0.f;
}

// ---------------------------------------------------------------------------
// Host-side launch helpers
// ---------------------------------------------------------------------------
#define TBH 256
#define GRID1(n) (((n) + TBH - 1) / TBH)

static void launch_gemm(hipStream_t st, const _Float16* A, int ldk,
                        const float* W, int K, int Kpad, int Nc,
                        const float* bias, float* C32, _Float16* C16, int ldc,
                        int M, int act, _Float16* wt) {
  prep_wt_kernel<<<GRID1(Nc * Kpad), TBH, 0, st>>>(W, K, Nc, Kpad, wt);
  dim3 g(M / 64);
  int nt = (Nc + 15) >> 4;
  switch (nt) {
    case 1: gemm_h_kernel<1><<<g, 32 * GW, 0, st>>>(A, ldk, wt, bias, C32, C16, ldc, Kpad, Nc, act); break;
    case 2: gemm_h_kernel<2><<<g, 32 * GW, 0, st>>>(A, ldk, wt, bias, C32, C16, ldc, Kpad, Nc, act); break;
    case 3: gemm_h_kernel<3><<<g, 32 * GW, 0, st>>>(A, ldk, wt, bias, C32, C16, ldc, Kpad, Nc, act); break;
    case 4: gemm_h_kernel<4><<<g, 32 * GW, 0, st>>>(A, ldk, wt, bias, C32, C16, ldc, Kpad, Nc, act); break;
    case 5: gemm_h_kernel<5><<<g, 32 * GW, 0, st>>>(A, ldk, wt, bias, C32, C16, ldc, Kpad, Nc, act); break;
    default: gemm_h_kernel<6><<<g, 32 * GW, 0, st>>>(A, ldk, wt, bias, C32, C16, ldc, Kpad, Nc, act); break;
  }
}
static void launch_bn(hipStream_t st, float* Hm, int rows, int cols,
                      const float* g, const float* b, const float* w,
                      float* stats, int act, _Float16* out16) {
  int ns = 2 * cols + 1;
  zero_f32_kernel<<<GRID1(ns), TBH, 0, st>>>(stats, ns);
  int rpb = 512;
  bn_stats_kernel<<<(rows + rpb - 1) / rpb, cols, 0, st>>>(Hm, cols, rows, cols,
                                                           w, stats, rpb);
  bn_apply_kernel<<<GRID1(rows * cols), TBH, 0, st>>>(
      Hm, cols, rows, cols, g, b, stats, w ? -1.f : (float)rows, act, out16);
}

extern "C" void kernel_launch(void* const* d_in, const int* in_sizes, int n_in,
                              void* d_out, int out_size, void* d_ws, size_t ws_size,
                              hipStream_t stream) {
  (void)in_sizes; (void)n_in; (void)out_size; (void)ws_size;
  hipStream_t st = stream;
  const float* x = (const float*)d_in[0];
#define PW(i) ((const float*)d_in[(i)])

  // workspace bump allocator (256B aligned)
  char* base = (char*)d_ws;
  size_t off = 0;
  auto ALF = [&](size_t n) -> float* {
    float* p = (float*)(base + off); off += ((n * 4 + 255) & ~(size_t)255); return p; };
  auto ALI = [&](size_t n) -> int* {
    int* p = (int*)(base + off); off += ((n * 4 + 255) & ~(size_t)255); return p; };
  auto ALH = [&](size_t n) -> _Float16* {
    _Float16* p = (_Float16*)(base + off); off += ((n * 2 + 255) & ~(size_t)255); return p; };

  float* t0      = ALF((size_t)N_NODES * HDIM);
  float* t1      = ALF((size_t)N_NODES * HDIM);
  float* xemb    = ALF((size_t)N_NODES * HDIM);
  float* xcat    = ALF((size_t)N_NODES * HDIM);
  float* xprop   = ALF((size_t)N_NODES * HDIM);
  float* fcat    = ALF((size_t)N_NODES * (2 * HDIM + IDIM));
  float* sq      = ALF(N_NODES);
  float* stats   = ALF(2 * MDIM + 1);
  float* wgood   = ALF(NEDGE);
  float* pooled  = ALF((size_t)N_NODES * HDIM);
  float* mf      = ALF((size_t)NEDGE * HDIM);         // f32 edge buffer (BN)
  _Float16* xpad16 = ALH((size_t)N_NODES * 32);
  _Float16* t0h    = ALH((size_t)N_NODES * HDIM);
  _Float16* t1h    = ALH((size_t)N_NODES * HDIM);
  _Float16* hbuf   = ALH((size_t)N_NODES * HDIM);     // kNN image / misc
  _Float16* wt     = ALH((size_t)MDIM * 288);         // transposed weights
  _Float16* m1h    = ALH((size_t)NEDGE * MDIM);
  _Float16* m2h    = ALH((size_t)NEDGE * MDIM);
  _Float16* Aedh   = ALH((size_t)NEDGE * 288);
  int*   kidx    = ALI(NEDGE);
  int*   srcU    = ALI(NEDGE);
  int*   la      = ALI(N_NODES);
  int*   lb      = ALI(N_NODES);
  int*   present = ALI(N_NODES);
  int*   cum     = ALI(N_NODES);
  int*   cmap    = ALI(N_NODES);
  unsigned* enc  = (unsigned*)ALI((size_t)N_NODES * HDIM);

  float* outp     = (float*)d_out;
  float* o_out    = outp;                                  // [N,O]
  float* o_scores = o_out + (size_t)N_NODES * ODIM;        // [E,2]
  float* o_eix    = o_scores + (size_t)NEDGE * 2;          // [2,E]
  float* o_cmap   = o_eix + (size_t)2 * NEDGE;             // [N]
  float* o_props  = o_cmap + N_NODES;                      // [N,1]
  float* o_batch  = o_props + N_NODES;                     // [N]

  pad_x16_kernel<<<GRID1(N_NODES * 32), TBH, 0, st>>>(x, xpad16);

  // ---------------- inputnet: x_emb ----------------
  launch_gemm(st, xpad16, 32, PW(1), IDIM, 32, HDIM, PW(2), t0, nullptr, HDIM,
              N_NODES, ACT_ELU, wt);
  launch_bn(st, t0, N_NODES, HDIM, PW(3), PW(4), nullptr, stats, ACT_NONE, t0h);
  launch_gemm(st, t0h, HDIM, PW(5), HDIM, HDIM, HDIM, PW(6), xemb, nullptr,
              HDIM, N_NODES, ACT_ELU, wt);
  launch_bn(st, xemb, N_NODES, HDIM, PW(7), PW(8), nullptr, stats, ACT_NONE, nullptr);

  // ---------------- edgeconvs (x3, recompute kNN each) ----------------
  for (int i = 0; i < 3; ++i) {
    int pb = 9 + 8 * i;
    tof16_kernel<<<GRID1(N_NODES * 64), TBH, 0, st>>>(xemb, N_NODES, HDIM, 64, hbuf);
    rownorm_kernel<<<GRID1(N_NODES), TBH, 0, st>>>(xemb, HDIM, sq, N_NODES);
    knn_kernel<<<N_NODES / 16, 32, 0, st>>>(hbuf, sq, kidx, N_NODES);
    edge_gather_kernel<<<GRID1(NEDGE), TBH, 0, st>>>(xemb, HDIM, kidx, 0, Aedh, 2 * HDIM);
    launch_gemm(st, Aedh, 2 * HDIM, PW(pb + 0), 2 * HDIM, 2 * HDIM, MDIM,
                PW(pb + 1), nullptr, m1h, MDIM, NEDGE, ACT_ELU, wt);
    launch_gemm(st, m1h, MDIM, PW(pb + 2), MDIM, MDIM, MDIM, PW(pb + 3),
                nullptr, m2h, MDIM, NEDGE, ACT_ELU, wt);
    launch_gemm(st, m2h, MDIM, PW(pb + 4), MDIM, MDIM, HDIM, PW(pb + 5), mf,
                nullptr, HDIM, NEDGE, ACT_ELU, wt);
    launch_bn(st, mf, NEDGE, HDIM, PW(pb + 6), PW(pb + 7), nullptr, stats,
              ACT_NONE, nullptr);
    scatter_sum_res_kernel<<<GRID1(N_NODES * HDIM), TBH, 0, st>>>(xemb, mf, HDIM);
  }

  // ---------------- output head -> out, kNN in output space ----------------
  tof16_kernel<<<GRID1(N_NODES * 64), TBH, 0, st>>>(xemb, N_NODES, HDIM, 64, hbuf);
  launch_gemm(st, hbuf, HDIM, PW(33), HDIM, HDIM, HDIM, PW(34), t0, nullptr,
              HDIM, N_NODES, ACT_ELU, wt);
  launch_bn(st, t0, N_NODES, HDIM, PW(35), PW(36), nullptr, stats, ACT_NONE, t0h);
  launch_gemm(st, t0h, HDIM, PW(37), HDIM, HDIM, HDIM, PW(38), nullptr, t1h,
              HDIM, N_NODES, ACT_ELU, wt);
  launch_gemm(st, t1h, HDIM, PW(39), HDIM, HDIM, ODIM, PW(40), o_out, nullptr,
              ODIM, N_NODES, ACT_NONE, wt);
  tof16_kernel<<<GRID1(N_NODES * 64), TBH, 0, st>>>(o_out, N_NODES, ODIM, 64, hbuf);
  rownorm_kernel<<<GRID1(N_NODES), TBH, 0, st>>>(o_out, ODIM, sq, N_NODES);
  knn_kernel<<<N_NODES / 16, 32, 0, st>>>(hbuf, sq, kidx, N_NODES);

  // ---------------- x_cat = net_tanh(inputnet_cat, x) + x_emb ----------------
  launch_gemm(st, xpad16, 32, PW(41), IDIM, 32, HDIM, PW(42), nullptr, t0h,
              HDIM, N_NODES, ACT_TANH, wt);
  launch_gemm(st, t0h, HDIM, PW(43), HDIM, HDIM, HDIM, PW(44), t1, nullptr,
              HDIM, N_NODES, ACT_NONE, wt);
  launch_bn(st, t1, N_NODES, HDIM, PW(45), PW(46), nullptr, stats, ACT_TANH, t1h);
  launch_gemm(st, t1h, HDIM, PW(47), HDIM, HDIM, HDIM, PW(48), t0, nullptr,
              HDIM, N_NODES, ACT_NONE, wt);
  launch_bn(st, t0, N_NODES, HDIM, PW(49), PW(50), nullptr, stats, ACT_TANH, nullptr);
  add2_kernel<<<GRID1(N_NODES * HDIM), TBH, 0, st>>>(xcat, t0, xemb, N_NODES * HDIM);

  // ---------------- edgecatconvs (x6) ----------------
  for (int i = 0; i < 6; ++i) {
    int pb = 51 + 6 * i;
    concat3_kernel<<<GRID1(N_NODES), TBH, 0, st>>>(xcat, xemb, x, fcat);
    edge_gather_kernel<<<GRID1(NEDGE), TBH, 0, st>>>(fcat, 2 * HDIM + IDIM, kidx, 0, Aedh, 288);
    launch_gemm(st, Aedh, 288, PW(pb + 0), 2 * (2 * HDIM + IDIM), 288, MDIM,
                PW(pb + 1), nullptr, m1h, MDIM, NEDGE, ACT_ELU, wt);
    launch_gemm(st, m1h, MDIM, PW(pb + 2), MDIM, MDIM, HDIM, PW(pb + 3), mf,
                nullptr, HDIM, NEDGE, ACT_ELU, wt);
    launch_bn(st, mf, NEDGE, HDIM, PW(pb + 4), PW(pb + 5), nullptr, stats,
              ACT_NONE, nullptr);
    scatter_sum_res_kernel<<<GRID1(N_NODES * HDIM), TBH, 0, st>>>(xcat, mf, HDIM);
  }

  // ---------------- edge classifier ----------------
  edge_gather_kernel<<<GRID1(NEDGE), TBH, 0, st>>>(xcat, HDIM, kidx, 1, Aedh, 2 * HDIM);
  launch_gemm(st, Aedh, 2 * HDIM, PW(87), 2 * HDIM, 2 * HDIM, HDIM, PW(88), mf,
              nullptr, HDIM, NEDGE, ACT_ELU, wt);
  launch_bn(st, mf, NEDGE, HDIM, PW(89), PW(90), nullptr, stats, ACT_NONE, m1h);
  launch_gemm(st, m1h, HDIM, PW(91), HDIM, HDIM, HDIM / 2, PW(92), mf, nullptr,
              HDIM / 2, NEDGE, ACT_ELU, wt);
  launch_bn(st, mf, NEDGE, HDIM / 2, PW(93), PW(94), nullptr, stats, ACT_NONE, m2h);
  launch_gemm(st, m2h, HDIM / 2, PW(95), HDIM / 2, HDIM / 2, 2, PW(96),
              o_scores, nullptr, 2, NEDGE, ACT_NONE, wt);
  good_kernel<<<GRID1(NEDGE), TBH, 0, st>>>(o_scores, kidx, wgood, srcU, o_eix);

  // ---------------- connected components + relabel ----------------
  init_labels_kernel<<<GRID1(N_NODES), TBH, 0, st>>>(la, N_NODES);
  for (int it = 0; it < 16; ++it) {
    relax_kernel<<<GRID1(NEDGE), TBH, 0, st>>>(la, srcU);
    jump_kernel<<<GRID1(N_NODES), TBH, 0, st>>>(la, lb, N_NODES);
    jump_kernel<<<GRID1(N_NODES), TBH, 0, st>>>(lb, la, N_NODES);
  }
  zero_f32_kernel<<<GRID1(N_NODES), TBH, 0, st>>>((float*)present, N_NODES);
  mark_kernel<<<GRID1(N_NODES), TBH, 0, st>>>(la, present, N_NODES);
  scan_kernel<<<1, 1, 0, st>>>(present, cum, N_NODES);
  cmap_kernel<<<GRID1(N_NODES), TBH, 0, st>>>(la, cum, cmap, o_cmap, N_NODES);

  // ---------------- x_prop = net_tanh(inputnet_prop, x) + x_emb ----------------
  launch_gemm(st, xpad16, 32, PW(97), IDIM, 32, HDIM, PW(98), nullptr, t0h,
              HDIM, N_NODES, ACT_TANH, wt);
  launch_gemm(st, t0h, HDIM, PW(99), HDIM, HDIM, HDIM, PW(100), t1, nullptr,
              HDIM, N_NODES, ACT_NONE, wt);
  launch_bn(st, t1, N_NODES, HDIM, PW(101), PW(102), nullptr, stats, ACT_TANH, t1h);
  launch_gemm(st, t1h, HDIM, PW(103), HDIM, HDIM, HDIM, PW(104), t0, nullptr,
              HDIM, N_NODES, ACT_NONE, wt);
  launch_bn(st, t0, N_NODES, HDIM, PW(105), PW(106), nullptr, stats, ACT_TANH, nullptr);
  add2_kernel<<<GRID1(N_NODES * HDIM), TBH, 0, st>>>(xprop, t0, xemb, N_NODES * HDIM);

  // ---------------- propertyconvs (x3, masked BN + masked max) ----------------
  for (int i = 0; i < 3; ++i) {
    int pb = 107 + 6 * i;
    concat3_kernel<<<GRID1(N_NODES), TBH, 0, st>>>(xprop, xemb, x, fcat);
    edge_gather_kernel<<<GRID1(NEDGE), TBH, 0, st>>>(fcat, 2 * HDIM + IDIM, kidx, 0, Aedh, 288);
    launch_gemm(st, Aedh, 288, PW(pb + 0), 2 * (2 * HDIM + IDIM), 288, MDIM,
                PW(pb + 1), nullptr, m1h, MDIM, NEDGE, ACT_ELU, wt);
    launch_gemm(st, m1h, MDIM, PW(pb + 2), MDIM, MDIM, HDIM, PW(pb + 3), mf,
                nullptr, HDIM, NEDGE, ACT_ELU, wt);
    launch_bn(st, mf, NEDGE, HDIM, PW(pb + 4), PW(pb + 5), wgood, stats,
              ACT_NONE, nullptr);
    prop_max_res_kernel<<<GRID1(N_NODES * HDIM), TBH, 0, st>>>(xprop, mf, HDIM, wgood);
  }

  // ---------------- cluster max-pool + property predictor ----------------
  encinit_kernel<<<GRID1(N_NODES * HDIM), TBH, 0, st>>>(enc, N_NODES * HDIM);
  pool_scatter_kernel<<<GRID1(N_NODES * HDIM), TBH, 0, st>>>(xprop, cmap, enc, N_NODES * HDIM);
  pool_decode_kernel<<<GRID1(N_NODES * HDIM), TBH, 0, st>>>(enc, pooled, N_NODES * HDIM);
  tof16_kernel<<<GRID1(N_NODES * 64), TBH, 0, st>>>(pooled, N_NODES, HDIM, 64, hbuf);
  launch_gemm(st, hbuf, HDIM, PW(125), HDIM, HDIM, HDIM, PW(126), nullptr, t0h,
              HDIM, N_NODES, ACT_ELU, wt);
  launch_gemm(st, t0h, HDIM, PW(127), HDIM, HDIM, HDIM / 2, PW(128), nullptr,
              t1h, HDIM / 2, N_NODES, ACT_ELU, wt);
  launch_gemm(st, t1h, HDIM / 2, PW(129), HDIM / 2, HDIM / 2, 1, PW(130),
              o_props, nullptr, 1, N_NODES, ACT_NONE, wt);
  zero_f32_kernel<<<GRID1(N_NODES), TBH, 0, st>>>(o_batch, N_NODES);
}
